// MultiHeadAttention_35888746725399
// MI455X (gfx1250) — matrix operand order repaired and checked by reading, not compiled, not run on hardware
//
#include <hip/hip_runtime.h>
#include <hip/hip_bf16.h>
#include <math.h>

// MultiHeadAttention forward for MI455X (gfx1250), bf16 WMMA path.
// B=4, T=2048, H=16, D=64, C=1024.

#define HH 16
#define DD 64
#define CC 1024
#define TT 2048
#define BB 4
#define BT (BB*TT)

typedef __attribute__((ext_vector_type(16))) __bf16 v16bf;
typedef __attribute__((ext_vector_type(8)))  float  v8f;
typedef __attribute__((ext_vector_type(4)))  int    vi4;
typedef unsigned short u16;
typedef unsigned int   u32;

__device__ __forceinline__ u16 f2bf(float f) {
  u32 u = __float_as_uint(f);
  u32 r = u + 0x7FFFu + ((u >> 16) & 1u);   // round-to-nearest-even
  return (u16)(r >> 16);
}

// Load a 16-element bf16 fragment as two 16-byte chunks (p0 -> elems 0..7, p1 -> 8..15).
__device__ __forceinline__ v16bf ld_frag(const u16* p0, const u16* p1) {
  union { v16bf v; vi4 q[2]; } u;
  u.q[0] = *(const vi4*)p0;
  u.q[1] = *(const vi4*)p1;
  return u.v;
}

__device__ __forceinline__ v8f wmma_bf16(v16bf a, v16bf b, v8f c) {
  return __builtin_amdgcn_wmma_f32_16x16x32_bf16(false, a, false, b, (short)0, c, false, false);
}

// ---------------- conversion kernels ----------------

__global__ __launch_bounds__(256) void k_cvt_bf16(const float* __restrict__ src,
                                                  u16* __restrict__ dst, int n) {
  int i = blockIdx.x * blockDim.x + threadIdx.x;
  int stride = gridDim.x * blockDim.x;
  for (; i < n; i += stride) dst[i] = f2bf(src[i]);
}

// Wq/Wk/Wv [H,C,D] fp32 -> wt [3][H][D][C] bf16 (transposed: contiguous in C for B-fragments)
__global__ __launch_bounds__(256) void k_cvt_wqkv_t(const float* __restrict__ Wq,
                                                    const float* __restrict__ Wk,
                                                    const float* __restrict__ Wv,
                                                    u16* __restrict__ wt) {
  const int per = HH * CC * DD;
  int i = blockIdx.x * blockDim.x + threadIdx.x;
  int stride = gridDim.x * blockDim.x;
  for (; i < 3 * per; i += stride) {
    int which = i / per;
    int rem = i - which * per;
    int h = rem / (CC * DD);
    int cd = rem - h * (CC * DD);
    int c = cd / DD;
    int d = cd - c * DD;
    const float* W = (which == 0) ? Wq : (which == 1) ? Wk : Wv;
    float v = W[(size_t)(h * CC + c) * DD + d];
    wt[((size_t)(which * HH + h) * DD + d) * CC + c] = f2bf(v);
  }
}

// ---------------- QKV projection GEMM ----------------
// grid: (BT/256, H, 3), 256 threads (8 waves). Each wave: 32 rows x 64 cols (one head).
// Dynamic double-buffered K loop (bounded scheduling region -> no spills).
// q,k stored [B,H,T,D]; v stored transposed [B,H,D,T].

__global__ __launch_bounds__(256) void k_qkv_gemm(const u16* __restrict__ xb,
                                                  const u16* __restrict__ wt,
                                                  u16* __restrict__ qo, u16* __restrict__ ko,
                                                  u16* __restrict__ vto) {
  int lane = threadIdx.x & 31;
  int wave = threadIdx.x >> 5;
  int half = lane >> 4, l15 = lane & 15;
  int which = blockIdx.z;
  int h = blockIdx.y;
  int rowbase = blockIdx.x * 256 + wave * 32;

  v8f acc[2][4] = {};
  const u16* arow[2];
  arow[0] = xb + (size_t)(rowbase + l15) * CC + half * 8;
  arow[1] = xb + (size_t)(rowbase + 16 + l15) * CC + half * 8;
  const u16* bbase[4];
#pragma unroll
  for (int f = 0; f < 4; ++f) {
    int d = f * 16 + l15;
    bbase[f] = wt + ((size_t)(which * HH + h) * DD + d) * CC + half * 8;
  }

  v16bf a0[2], a1[2], b0[4], b1[4];
#pragma unroll
  for (int i = 0; i < 2; ++i) a0[i] = ld_frag(arow[i], arow[i] + 16);
#pragma unroll
  for (int f = 0; f < 4; ++f) b0[f] = ld_frag(bbase[f], bbase[f] + 16);

#pragma unroll 1
  for (int kc = 0; kc < CC - 64; kc += 64) {
#pragma unroll
    for (int i = 0; i < 2; ++i) a1[i] = ld_frag(arow[i] + kc + 32, arow[i] + kc + 48);
#pragma unroll
    for (int f = 0; f < 4; ++f) b1[f] = ld_frag(bbase[f] + kc + 32, bbase[f] + kc + 48);
#pragma unroll
    for (int i = 0; i < 2; ++i)
#pragma unroll
      for (int f = 0; f < 4; ++f) acc[i][f] = wmma_bf16(a0[i], b0[f], acc[i][f]);
#pragma unroll
    for (int i = 0; i < 2; ++i) a0[i] = ld_frag(arow[i] + kc + 64, arow[i] + kc + 80);
#pragma unroll
    for (int f = 0; f < 4; ++f) b0[f] = ld_frag(bbase[f] + kc + 64, bbase[f] + kc + 80);
#pragma unroll
    for (int i = 0; i < 2; ++i)
#pragma unroll
      for (int f = 0; f < 4; ++f) acc[i][f] = wmma_bf16(a1[i], b1[f], acc[i][f]);
  }
  {
    const int kc = CC - 64;
#pragma unroll
    for (int i = 0; i < 2; ++i) a1[i] = ld_frag(arow[i] + kc + 32, arow[i] + kc + 48);
#pragma unroll
    for (int f = 0; f < 4; ++f) b1[f] = ld_frag(bbase[f] + kc + 32, bbase[f] + kc + 48);
#pragma unroll
    for (int i = 0; i < 2; ++i)
#pragma unroll
      for (int f = 0; f < 4; ++f) acc[i][f] = wmma_bf16(a0[i], b0[f], acc[i][f]);
#pragma unroll
    for (int i = 0; i < 2; ++i)
#pragma unroll
      for (int f = 0; f < 4; ++f) acc[i][f] = wmma_bf16(a1[i], b1[f], acc[i][f]);
  }

#pragma unroll
  for (int i = 0; i < 2; ++i)
#pragma unroll
    for (int f = 0; f < 4; ++f) {
      int d = f * 16 + l15;
#pragma unroll
      for (int r = 0; r < 8; ++r) {
        int row = rowbase + i * 16 + r + half * 8;
        int bb = row >> 11;              // /TT
        int t = row & (TT - 1);
        u16 val = f2bf(acc[i][f][r]);
        if (which == 2) {
          vto[((size_t)(bb * HH + h) * DD + d) * TT + t] = val;
        } else {
          u16* dst = (which == 0) ? qo : ko;
          dst[((size_t)(bb * HH + h) * TT + t) * DD + d] = val;
        }
      }
    }
}

// ---------------- flash attention ----------------
// grid: B*H*(T/32)/8 blocks, 256 threads. Each wave: one 32-row Q tile (2 row-frags),
// online softmax over causal 64-wide KV blocks. K and V B-fragments are ping-pong
// pipelined per 16-column sub-block to bound live registers (no spills).

__global__ __launch_bounds__(256) void k_attn(const u16* __restrict__ q,
                                              const u16* __restrict__ k,
                                              const u16* __restrict__ vt,
                                              u16* __restrict__ attn) {
  __shared__ __align__(16) u16 plds[8][32 * 64];
  int lane = threadIdx.x & 31;
  int wave = threadIdx.x >> 5;
  int half = lane >> 4, l15 = lane & 15;

  int g = blockIdx.x * 8 + wave;       // global Q-tile id
  int bb = g >> 10;                    // / (H * T/32)
  int rem = g & 1023;
  int h = rem >> 6;
  int qt = rem & 63;
  int tq = qt * 32;

  const size_t bh = (size_t)(bb * HH + h);
  const u16* qbase = q + bh * TT * DD;
  const u16* kbase = k + bh * TT * DD;
  const u16* vbase = vt + bh * DD * TT;

  // Q A-fragments: 2 row-frags x (K = D = 64 -> two K=32 frags)
  v16bf aq[2][2];
#pragma unroll
  for (int i = 0; i < 2; ++i) {
    const u16* qrow = qbase + (size_t)(tq + i * 16 + l15) * DD + half * 8;
    aq[i][0] = ld_frag(qrow, qrow + 16);
    aq[i][1] = ld_frag(qrow + 32, qrow + 32 + 16);
  }

  v8f o[2][4] = {};
  float m_run[2][8], l_run[2][8];
#pragma unroll
  for (int i = 0; i < 2; ++i)
#pragma unroll
    for (int r = 0; r < 8; ++r) { m_run[i][r] = -INFINITY; l_run[i][r] = 0.f; }

  const float scale = 0.03125f;        // 1/sqrt(C) = 1/32
  u16* pl = plds[wave];

#pragma unroll 1
  for (int s0 = 0; s0 <= tq; s0 += 64) {
    // warm next block's K/V lines one block ahead (global_prefetch_b8)
    if (s0 + 64 <= tq) {
      __builtin_prefetch(kbase + (size_t)(s0 + 64 + 2 * lane) * DD, 0, 1);
      __builtin_prefetch(vbase + (size_t)(2 * lane) * TT + s0 + 64, 0, 1);
    }

    // ---- scores: K fragments ping-pong pipelined over the 4 sub-blocks ----
    v8f sc[2][4];
    v16bf kfb[2][2];
    {
      const u16* krow = kbase + (size_t)(s0 + l15) * DD + half * 8;
      kfb[0][0] = ld_frag(krow, krow + 16);
      kfb[0][1] = ld_frag(krow + 32, krow + 32 + 16);
    }
#pragma unroll
    for (int f = 0; f < 4; ++f) {
      if (f < 3) {
        const u16* krow = kbase + (size_t)(s0 + (f + 1) * 16 + l15) * DD + half * 8;
        kfb[(f + 1) & 1][0] = ld_frag(krow, krow + 16);
        kfb[(f + 1) & 1][1] = ld_frag(krow + 32, krow + 32 + 16);
      }
#pragma unroll
      for (int i = 0; i < 2; ++i) {
        v8f c = {};
        c = wmma_bf16(aq[i][0], kfb[f & 1][0], c);
        c = wmma_bf16(aq[i][1], kfb[f & 1][1], c);
        sc[i][f] = c;
      }
    }

    bool full = (s0 + 63 <= tq);
#pragma unroll
    for (int i = 0; i < 2; ++i)
#pragma unroll
      for (int f = 0; f < 4; ++f) {
        int s = s0 + f * 16 + l15;
#pragma unroll
        for (int r = 0; r < 8; ++r) {
          float v = sc[i][f][r] * scale;
          if (!full) {
            int t = tq + i * 16 + r + half * 8;
            if (s > t) v = -INFINITY;
          }
          sc[i][f][r] = v;
        }
      }
    // online softmax per row-frag: half-wave shuffle reductions
#pragma unroll
    for (int i = 0; i < 2; ++i) {
#pragma unroll
      for (int r = 0; r < 8; ++r) {
        float mx = fmaxf(fmaxf(sc[i][0][r], sc[i][1][r]), fmaxf(sc[i][2][r], sc[i][3][r]));
        mx = fmaxf(mx, __shfl_xor(mx, 1));
        mx = fmaxf(mx, __shfl_xor(mx, 2));
        mx = fmaxf(mx, __shfl_xor(mx, 4));
        mx = fmaxf(mx, __shfl_xor(mx, 8));
        float mnew = fmaxf(m_run[i][r], mx);
        float alpha = __expf(m_run[i][r] - mnew);
        m_run[i][r] = mnew;
        float rs = 0.f;
#pragma unroll
        for (int f = 0; f < 4; ++f) {
          float p = __expf(sc[i][f][r] - mnew);
          rs += p;
          pl[(i * 16 + r + half * 8) * 64 + f * 16 + l15] = f2bf(p);
        }
        rs += __shfl_xor(rs, 1);
        rs += __shfl_xor(rs, 2);
        rs += __shfl_xor(rs, 4);
        rs += __shfl_xor(rs, 8);
        l_run[i][r] = l_run[i][r] * alpha + rs;
#pragma unroll
        for (int f = 0; f < 4; ++f) o[i][f][r] *= alpha;
      }
    }

    // first V fragment issued before the DS wait: overlaps the LDS round-trip
    v16bf vfb[2][2];
    {
      const u16* vrow = vbase + (size_t)l15 * TT + s0 + half * 8;
      vfb[0][0] = ld_frag(vrow, vrow + 16);
      vfb[0][1] = ld_frag(vrow + 32, vrow + 32 + 16);
    }

    asm volatile("s_wait_dscnt 0x0" ::: "memory");   // intra-wave DS write->read ordering
    // P as A-fragments from LDS
    v16bf ap[2][2];
#pragma unroll
    for (int i = 0; i < 2; ++i) {
      const u16* prow = pl + (i * 16 + l15) * 64 + half * 8;
      ap[i][0] = ld_frag(prow, prow + 16);
      ap[i][1] = ld_frag(prow + 32, prow + 32 + 16);
    }
    // ---- O += P @ V: V fragments ping-pong pipelined ----
#pragma unroll
    for (int f = 0; f < 4; ++f) {
      if (f < 3) {
        const u16* vrow = vbase + (size_t)((f + 1) * 16 + l15) * TT + s0 + half * 8;
        vfb[(f + 1) & 1][0] = ld_frag(vrow, vrow + 16);
        vfb[(f + 1) & 1][1] = ld_frag(vrow + 32, vrow + 32 + 16);
      }
#pragma unroll
      for (int i = 0; i < 2; ++i) {
        o[i][f] = wmma_bf16(ap[i][0], vfb[f & 1][0], o[i][f]);
        o[i][f] = wmma_bf16(ap[i][1], vfb[f & 1][1], o[i][f]);
      }
    }
  }
  // normalize and store concatenated-head layout [B,T,C]
#pragma unroll
  for (int i = 0; i < 2; ++i)
#pragma unroll
    for (int r = 0; r < 8; ++r) {
      float inv = 1.f / l_run[i][r];
      int t = tq + i * 16 + r + half * 8;
#pragma unroll
      for (int f = 0; f < 4; ++f) {
        int col = h * DD + f * 16 + l15;
        attn[((size_t)bb * TT + t) * CC + col] = f2bf(o[i][f][r] * inv);
      }
    }
}

// ---------------- output projection: y = attn @ Wp^T + bp ----------------
// grid: (BT/256, C/64), 256 threads, 32 rows x 64 cols per wave.
// B[k=c][n=j] = Wp[j][c] (already contiguous in K). Same pipeline as QKV GEMM.

__global__ __launch_bounds__(256) void k_proj(const u16* __restrict__ attn,
                                              const u16* __restrict__ wpb,
                                              const float* __restrict__ bp,
                                              float* __restrict__ out) {
  int lane = threadIdx.x & 31;
  int wave = threadIdx.x >> 5;
  int half = lane >> 4, l15 = lane & 15;
  int rowbase = blockIdx.x * 256 + wave * 32;
  int colbase = blockIdx.y * 64;

  v8f acc[2][4] = {};
  const u16* arow[2];
  arow[0] = attn + (size_t)(rowbase + l15) * CC + half * 8;
  arow[1] = attn + (size_t)(rowbase + 16 + l15) * CC + half * 8;
  const u16* bbase[4];
#pragma unroll
  for (int f = 0; f < 4; ++f) {
    int j = colbase + f * 16 + l15;
    bbase[f] = wpb + (size_t)j * CC + half * 8;
  }

  v16bf a0[2], a1[2], b0[4], b1[4];
#pragma unroll
  for (int i = 0; i < 2; ++i) a0[i] = ld_frag(arow[i], arow[i] + 16);
#pragma unroll
  for (int f = 0; f < 4; ++f) b0[f] = ld_frag(bbase[f], bbase[f] + 16);

#pragma unroll 1
  for (int kc = 0; kc < CC - 64; kc += 64) {
#pragma unroll
    for (int i = 0; i < 2; ++i) a1[i] = ld_frag(arow[i] + kc + 32, arow[i] + kc + 48);
#pragma unroll
    for (int f = 0; f < 4; ++f) b1[f] = ld_frag(bbase[f] + kc + 32, bbase[f] + kc + 48);
#pragma unroll
    for (int i = 0; i < 2; ++i)
#pragma unroll
      for (int f = 0; f < 4; ++f) acc[i][f] = wmma_bf16(a0[i], b0[f], acc[i][f]);
#pragma unroll
    for (int i = 0; i < 2; ++i) a0[i] = ld_frag(arow[i] + kc + 64, arow[i] + kc + 80);
#pragma unroll
    for (int f = 0; f < 4; ++f) b0[f] = ld_frag(bbase[f] + kc + 64, bbase[f] + kc + 80);
#pragma unroll
    for (int i = 0; i < 2; ++i)
#pragma unroll
      for (int f = 0; f < 4; ++f) acc[i][f] = wmma_bf16(a1[i], b1[f], acc[i][f]);
  }
  {
    const int kc = CC - 64;
#pragma unroll
    for (int i = 0; i < 2; ++i) a1[i] = ld_frag(arow[i] + kc + 32, arow[i] + kc + 48);
#pragma unroll
    for (int f = 0; f < 4; ++f) b1[f] = ld_frag(bbase[f] + kc + 32, bbase[f] + kc + 48);
#pragma unroll
    for (int i = 0; i < 2; ++i)
#pragma unroll
      for (int f = 0; f < 4; ++f) acc[i][f] = wmma_bf16(a0[i], b0[f], acc[i][f]);
#pragma unroll
    for (int i = 0; i < 2; ++i)
#pragma unroll
      for (int f = 0; f < 4; ++f) acc[i][f] = wmma_bf16(a1[i], b1[f], acc[i][f]);
  }

#pragma unroll
  for (int i = 0; i < 2; ++i)
#pragma unroll
    for (int f = 0; f < 4; ++f) {
      int j = colbase + f * 16 + l15;
      float bias = bp[j];
#pragma unroll
      for (int r = 0; r < 8; ++r) {
        int row = rowbase + i * 16 + r + half * 8;
        out[(size_t)row * CC + j] = acc[i][f][r] + bias;
      }
    }
}

// ---------------- host launch ----------------

extern "C" void kernel_launch(void* const* d_in, const int* in_sizes, int n_in,
                              void* d_out, int out_size, void* d_ws, size_t ws_size,
                              hipStream_t stream) {
  const float* x  = (const float*)d_in[0];
  const float* Wq = (const float*)d_in[1];
  const float* Wk = (const float*)d_in[2];
  const float* Wv = (const float*)d_in[3];
  const float* Wp = (const float*)d_in[4];
  const float* bp = (const float*)d_in[5];
  float* out = (float*)d_out;

  char* ws = (char*)d_ws;
  const size_t MB = 1024 * 1024;
  u16* xb   = (u16*)(ws);                 // 16 MB: x bf16 [BT, C]
  u16* wt   = (u16*)(ws + 16 * MB);       //  6 MB: qkv weights bf16 [3][H][D][C]
  u16* wpb  = (u16*)(ws + 23 * MB);       //  2 MB: Wp bf16 [C, C]
  u16* qb   = (u16*)(ws + 25 * MB);       // 16 MB: q bf16 [B,H,T,D]
  u16* kb   = (u16*)(ws + 41 * MB);       // 16 MB: k bf16 [B,H,T,D]
  u16* vtb  = (u16*)(ws + 57 * MB);       // 16 MB: v bf16 transposed [B,H,D,T]
  u16* attn = (u16*)(ws + 73 * MB);       // 16 MB: attention output bf16 [B,T,C]

  hipLaunchKernelGGL(k_cvt_bf16,   dim3(2048), dim3(256), 0, stream, x, xb, BT * CC);
  hipLaunchKernelGGL(k_cvt_wqkv_t, dim3(2048), dim3(256), 0, stream, Wq, Wk, Wv, wt);
  hipLaunchKernelGGL(k_cvt_bf16,   dim3(1024), dim3(256), 0, stream, Wp, wpb, CC * CC);
  hipLaunchKernelGGL(k_qkv_gemm,   dim3(BT / 256, HH, 3), dim3(256), 0, stream, xb, wt, qb, kb, vtb);
  hipLaunchKernelGGL(k_attn,       dim3(BB * HH * (TT / 32) / 8), dim3(256), 0, stream, qb, kb, vtb, attn);
  hipLaunchKernelGGL(k_proj,       dim3(BT / 256, CC / 64), dim3(256), 0, stream, attn, wpb, bp, out);
}
